// SelfAttention_50525995270194
// MI455X (gfx1250) — matrix operand order repaired and checked by reading, not compiled, not run on hardware
//
#include <hip/hip_runtime.h>
#include <hip/hip_bf16.h>

// ---------------------------------------------------------------------------
// MI455X (gfx1250) wave32 WMMA implementation of causal self-attention block.
// ---------------------------------------------------------------------------

#define DM   1024
#define NH   16
#define DK   64
#define SEQL 4096

typedef __attribute__((ext_vector_type(16))) _Float16 v16h;
typedef __attribute__((ext_vector_type(8)))  _Float16 v8h;
typedef __attribute__((ext_vector_type(8)))  float    v8f;

__device__ __forceinline__ v8f wmma_f16(v16h a, v16h b, v8f c) {
  // (neg_a, A, neg_b, B, c_mod, C, reuse_a, reuse_b)
  return __builtin_amdgcn_wmma_f32_16x16x32_f16(false, a, false, b, (short)0, c,
                                                false, false);
}

union AFrag { v16h v; v8h h[2]; };

// ---------------------------------------------------------------------------
// 1) LayerNorm: h[4096,1024] f32 -> hn f16
// ---------------------------------------------------------------------------
__global__ void ln_f16_kernel(const float* __restrict__ x,
                              const float* __restrict__ g,
                              const float* __restrict__ b,
                              _Float16* __restrict__ out) {
  __shared__ float s1[256];
  __shared__ float s2[256];
  const int row = blockIdx.x;
  const float* xr = x + (size_t)row * DM;
  float v[4];
  float a = 0.f, a2 = 0.f;
#pragma unroll
  for (int i = 0; i < 4; ++i) {
    float t = xr[threadIdx.x + 256 * i];
    v[i] = t;
    a += t;
    a2 += t * t;
  }
  s1[threadIdx.x] = a;
  s2[threadIdx.x] = a2;
  __syncthreads();
  for (int off = 128; off > 0; off >>= 1) {
    if (threadIdx.x < (unsigned)off) {
      s1[threadIdx.x] += s1[threadIdx.x + off];
      s2[threadIdx.x] += s2[threadIdx.x + off];
    }
    __syncthreads();
  }
  const float mu   = s1[0] * (1.f / DM);
  const float var  = s2[0] * (1.f / DM) - mu * mu;
  const float rstd = rsqrtf(var + 1e-5f);
#pragma unroll
  for (int i = 0; i < 4; ++i) {
    int c = threadIdx.x + 256 * i;
    out[(size_t)row * DM + c] = (_Float16)((v[i] - mu) * rstd * g[c] + b[c]);
  }
}

// ---------------------------------------------------------------------------
// 2) Weight transpose + f32->f16:  Wt[n][k] = (f16) W[k][n]   (1024x1024)
// ---------------------------------------------------------------------------
__global__ void wtrans_kernel(const float* __restrict__ W,
                              _Float16* __restrict__ Wt) {
  int idx = blockIdx.x * 256 + threadIdx.x;  // out index, k fastest
  int kk = idx & (DM - 1);
  int nn = idx >> 10;
  Wt[idx] = (_Float16)W[(size_t)kk * DM + nn];
}

// ---------------------------------------------------------------------------
// 3) Generic WMMA GEMM:  C[M,N] = A[M,K](f16) * Bt[N,K](f16)^T + bias
//    4 waves / block; wave w computes a 32x64 tile (2 M-subtiles x 4 N-tiles).
//    All 6 fragments of a k-step are loaded into distinct registers BEFORE the
//    8 WMMAs, with loop-invariant row pointers (k0 folds into the 24-bit
//    instruction offset), so no VALU lands between WMMAs (avoids hazard NOPs)
//    and the scheduler can overlap next-step loads with current WMMAs.
// ---------------------------------------------------------------------------
__global__ void gemm_wmma_kernel(const _Float16* __restrict__ A,
                                 const _Float16* __restrict__ Bt,
                                 const float* __restrict__ bias,
                                 const float* __restrict__ residual,
                                 _Float16* __restrict__ outH,
                                 float* __restrict__ outF,
                                 int M, int N, int K, int transposedStore) {
  const int wave  = threadIdx.x >> 5;
  const int lane  = threadIdx.x & 31;
  const int lhalf = lane >> 4;
  const int lmod  = lane & 15;
  const int mbase = blockIdx.x * 128 + wave * 32;  // 32 rows per wave
  const int nbase = blockIdx.y * 64;

  v8f acc[2][4];
#pragma unroll
  for (int m = 0; m < 2; ++m)
#pragma unroll
    for (int t = 0; t < 4; ++t)
#pragma unroll
      for (int r = 0; r < 8; ++r) acc[m][t][r] = 0.f;

  // Loop-invariant base pointers.
  const _Float16* a0lo = A + (size_t)(mbase + lmod) * K + lhalf * 16;
  const _Float16* a0hi = a0lo + 8;
  const _Float16* a1lo = A + (size_t)(mbase + 16 + lmod) * K + lhalf * 16;
  const _Float16* a1hi = a1lo + 8;
  const _Float16* bp0 = Bt + (size_t)(nbase + 0 * 16 + lmod) * K + lhalf * 16;
  const _Float16* bp1 = Bt + (size_t)(nbase + 1 * 16 + lmod) * K + lhalf * 16;
  const _Float16* bp2 = Bt + (size_t)(nbase + 2 * 16 + lmod) * K + lhalf * 16;
  const _Float16* bp3 = Bt + (size_t)(nbase + 3 * 16 + lmod) * K + lhalf * 16;

#pragma unroll 4
  for (int k0 = 0; k0 < K; k0 += 32) {
    AFrag ua0, ua1;
    ua0.h[0] = *(const v8h*)(a0lo + k0);
    ua0.h[1] = *(const v8h*)(a0hi + k0);
    ua1.h[0] = *(const v8h*)(a1lo + k0);
    ua1.h[1] = *(const v8h*)(a1hi + k0);
    v16h b0 = *(const v16h*)(bp0 + k0);
    v16h b1 = *(const v16h*)(bp1 + k0);
    v16h b2 = *(const v16h*)(bp2 + k0);
    v16h b3 = *(const v16h*)(bp3 + k0);
    acc[0][0] = wmma_f16(ua0.v, b0, acc[0][0]);
    acc[1][0] = wmma_f16(ua1.v, b0, acc[1][0]);
    acc[0][1] = wmma_f16(ua0.v, b1, acc[0][1]);
    acc[1][1] = wmma_f16(ua1.v, b1, acc[1][1]);
    acc[0][2] = wmma_f16(ua0.v, b2, acc[0][2]);
    acc[1][2] = wmma_f16(ua1.v, b2, acc[1][2]);
    acc[0][3] = wmma_f16(ua0.v, b3, acc[0][3]);
    acc[1][3] = wmma_f16(ua1.v, b3, acc[1][3]);
  }

#pragma unroll
  for (int m = 0; m < 2; ++m) {
#pragma unroll
    for (int t = 0; t < 4; ++t) {
#pragma unroll
      for (int r = 0; r < 8; ++r) {
        int row = mbase + m * 16 + r + 8 * lhalf;
        int col = nbase + t * 16 + lmod;
        float val = acc[m][t][r] + bias[col];
        if (outF) {
          float res = residual ? residual[(size_t)row * N + col] : 0.f;
          outF[(size_t)row * N + col] = val + res;
        } else if (transposedStore) {
          outH[(size_t)col * M + row] = (_Float16)val;
        } else {
          outH[(size_t)row * N + col] = (_Float16)val;
        }
      }
    }
  }
}

// ---------------------------------------------------------------------------
// 4) RoPE in place on q,k (f16, layout [s][h*64+d]); pairs (d, d+32), d<32.
// ---------------------------------------------------------------------------
__global__ void rope_kernel(_Float16* __restrict__ q, _Float16* __restrict__ k) {
  int idx = blockIdx.x * 256 + threadIdx.x;  // SEQL*NH*32 threads
  int d  = idx & 31;
  int hd = (idx >> 5) & (NH - 1);
  int s  = idx >> 9;
  float theta = __powf(10000.f, -(float)(2 * d) / (float)DK);
  float ang = (float)s * theta;
  float sn, cs;
  __sincosf(ang, &sn, &cs);
  size_t base = (size_t)s * DM + hd * DK + d;
  float q0 = (float)q[base], q1 = (float)q[base + 32];
  q[base]      = (_Float16)(q0 * cs - q1 * sn);
  q[base + 32] = (_Float16)(q1 * cs + q0 * sn);
  float k0 = (float)k[base], k1 = (float)k[base + 32];
  k[base]      = (_Float16)(k0 * cs - k1 * sn);
  k[base + 32] = (_Float16)(k1 * cs + k0 * sn);
}

// ---------------------------------------------------------------------------
// 5) Flash attention. grid = (SEQL/64, NH), block = 128 (4 waves).
//    Wave w owns 16 query rows. q,k: [s][h*64+d] f16. vT: [h][d][s] f16.
// ---------------------------------------------------------------------------
__global__ void attn_kernel(const _Float16* __restrict__ q,
                            const _Float16* __restrict__ k,
                            const _Float16* __restrict__ vT,
                            _Float16* __restrict__ o) {
  __shared__ __align__(16) _Float16 pbuf[4][16][64];
  const int wave  = threadIdx.x >> 5;
  const int lane  = threadIdx.x & 31;
  const int lhalf = lane >> 4;
  const int lmod  = lane & 15;
  const int head  = blockIdx.y;
  const int qrow0 = blockIdx.x * 64 + wave * 16;

  // Q fragments (16 rows x 64 d) -> two 16x32 A-frags
  AFrag aq[2];
  const _Float16* qrow = q + (size_t)(qrow0 + lmod) * DM + head * DK;
#pragma unroll
  for (int kk = 0; kk < 2; ++kk) {
    aq[kk].h[0] = *(const v8h*)(qrow + kk * 32 + lhalf * 16);
    aq[kk].h[1] = *(const v8h*)(qrow + kk * 32 + lhalf * 16 + 8);
  }

  // Loop-invariant K/V fragment pointers (jbase advances by 64 elements).
  const _Float16* kp[4];
  const _Float16* vp[4];
#pragma unroll
  for (int t = 0; t < 4; ++t) {
    kp[t] = k + (size_t)(t * 16 + lmod) * DM + head * DK + lhalf * 16;
    vp[t] = vT + (size_t)(head * DK + t * 16 + lmod) * SEQL + lhalf * 16;
  }

  float m_i[8], l_i[8];
  v8f oacc[4];
#pragma unroll
  for (int r = 0; r < 8; ++r) { m_i[r] = -3.0e38f; l_i[r] = 0.f; }
#pragma unroll
  for (int t = 0; t < 4; ++t)
#pragma unroll
    for (int r = 0; r < 8; ++r) oacc[t][r] = 0.f;

  const int nkb = blockIdx.x + 1;
  for (int kb = 0; kb < nkb; ++kb) {
    const int jbase = kb * 64;

    // Prefetch next key block's K rows and V rows (global_prefetch_b8).
    if (kb + 1 < nkb) {
      const int jn = jbase + 64;
      __builtin_prefetch(k + (size_t)(jn + lane) * DM + head * DK, 0, 1);
      __builtin_prefetch(k + (size_t)(jn + 32 + lane) * DM + head * DK, 0, 1);
      __builtin_prefetch(vT + (size_t)(head * DK + 2 * lane) * SEQL + jn, 0, 1);
      __builtin_prefetch(vT + (size_t)(head * DK + 2 * lane + 1) * SEQL + jn, 0,
                         1);
    }

    // ---- S = Q * K^T : load all 8 K fragments, then 8 WMMAs -------------
    v16h kf[4][2];
#pragma unroll
    for (int t = 0; t < 4; ++t) {
      const _Float16* krow = kp[t] + (size_t)jbase * DM;
      kf[t][0] = *(const v16h*)(krow);
      kf[t][1] = *(const v16h*)(krow + 32);
    }
    v8f sacc[4];
#pragma unroll
    for (int t = 0; t < 4; ++t)
#pragma unroll
      for (int r = 0; r < 8; ++r) sacc[t][r] = 0.f;
#pragma unroll
    for (int t = 0; t < 4; ++t) {
      sacc[t] = wmma_f16(aq[0].v, kf[t][0], sacc[t]);
      sacc[t] = wmma_f16(aq[1].v, kf[t][1], sacc[t]);
    }

    // ---- online softmax --------------------------------------------------
    const bool diag = (kb == blockIdx.x);
    float mnew[8];
#pragma unroll
    for (int r = 0; r < 8; ++r) {
      int row = qrow0 + r + 8 * lhalf;
      float mx = -3.0e38f;
#pragma unroll
      for (int t = 0; t < 4; ++t) {
        float sv = sacc[t][r] * 0.125f;  // 1/sqrt(64)
        int col = jbase + t * 16 + lmod;
        if (diag && col > row) sv = -3.0e38f;
        sacc[t][r] = sv;
        mx = fmaxf(mx, sv);
      }
#pragma unroll
      for (int off = 1; off <= 8; off <<= 1)
        mx = fmaxf(mx, __shfl_xor(mx, off, 32));
      mnew[r] = fmaxf(m_i[r], mx);
    }

    // P = exp(S - m), row sums, stage P in LDS, rescale O
#pragma unroll
    for (int r = 0; r < 8; ++r) {
      int prow = r + 8 * lhalf;
      float rs = 0.f;
#pragma unroll
      for (int t = 0; t < 4; ++t) {
        float p = __expf(sacc[t][r] - mnew[r]);
        rs += p;
        pbuf[wave][prow][t * 16 + lmod] = (_Float16)p;
      }
#pragma unroll
      for (int off = 1; off <= 8; off <<= 1) rs += __shfl_xor(rs, off, 32);
      float scale = __expf(m_i[r] - mnew[r]);
      l_i[r] = l_i[r] * scale + rs;
      m_i[r] = mnew[r];
#pragma unroll
      for (int t = 0; t < 4; ++t) oacc[t][r] *= scale;
    }

    // ---- O += P * V : reload P via LDS, load all 8 V frags, 8 WMMAs -----
    AFrag ap[2];
#pragma unroll
    for (int kk = 0; kk < 2; ++kk) {
      const _Float16* pr = &pbuf[wave][lmod][0];
      ap[kk].h[0] = *(const v8h*)(pr + kk * 32 + lhalf * 16);
      ap[kk].h[1] = *(const v8h*)(pr + kk * 32 + lhalf * 16 + 8);
    }
    v16h vf[4][2];
#pragma unroll
    for (int t = 0; t < 4; ++t) {
      const _Float16* vrow = vp[t] + jbase;
      vf[t][0] = *(const v16h*)(vrow);
      vf[t][1] = *(const v16h*)(vrow + 32);
    }
#pragma unroll
    for (int t = 0; t < 4; ++t) {
      oacc[t] = wmma_f16(ap[0].v, vf[t][0], oacc[t]);
      oacc[t] = wmma_f16(ap[1].v, vf[t][1], oacc[t]);
    }
  }

  // normalize + store f16 [s][h*64+d]
#pragma unroll
  for (int r = 0; r < 8; ++r) {
    float inv = 1.f / l_i[r];
    int row = qrow0 + r + 8 * lhalf;
#pragma unroll
    for (int t = 0; t < 4; ++t) {
      o[(size_t)row * DM + head * DK + t * 16 + lmod] =
          (_Float16)(oacc[t][r] * inv);
    }
  }
}

// ---------------------------------------------------------------------------
// Host-side launch
// ---------------------------------------------------------------------------
extern "C" void kernel_launch(void* const* d_in, const int* in_sizes, int n_in,
                              void* d_out, int out_size, void* d_ws,
                              size_t ws_size, hipStream_t stream) {
  (void)in_sizes; (void)n_in; (void)out_size; (void)ws_size;
  const float* h     = (const float*)d_in[0];
  const float* gamma = (const float*)d_in[1];
  const float* beta  = (const float*)d_in[2];
  const float* Wq    = (const float*)d_in[3];
  const float* bq    = (const float*)d_in[4];
  const float* Wk    = (const float*)d_in[5];
  const float* bk    = (const float*)d_in[6];
  const float* Wv    = (const float*)d_in[7];
  const float* bv    = (const float*)d_in[8];
  const float* Wo    = (const float*)d_in[9];
  const float* bo    = (const float*)d_in[10];
  float* out = (float*)d_out;

  char* ws = (char*)d_ws;
  const size_t MB = 1u << 20;
  _Float16* hn  = (_Float16*)(ws + 0 * MB);    // 8 MiB  [s][dm]
  _Float16* qr  = (_Float16*)(ws + 8 * MB);    // 8 MiB  [s][dm]
  _Float16* kr  = (_Float16*)(ws + 16 * MB);   // 8 MiB  [s][dm]
  _Float16* vT  = (_Float16*)(ws + 24 * MB);   // 8 MiB  [h][d][s]
  _Float16* ob  = (_Float16*)(ws + 32 * MB);   // 8 MiB  [s][dm]
  _Float16* WqT = (_Float16*)(ws + 40 * MB);   // 2 MiB each
  _Float16* WkT = (_Float16*)(ws + 42 * MB);
  _Float16* WvT = (_Float16*)(ws + 44 * MB);
  _Float16* WoT = (_Float16*)(ws + 46 * MB);   // total 48 MiB

  // 1) LayerNorm
  ln_f16_kernel<<<SEQL, 256, 0, stream>>>(h, gamma, beta, hn);

  // 2) Weight transposes (f32 -> f16, [n][k])
  const int wtBlocks = (DM * DM) / 256;
  wtrans_kernel<<<wtBlocks, 256, 0, stream>>>(Wq, WqT);
  wtrans_kernel<<<wtBlocks, 256, 0, stream>>>(Wk, WkT);
  wtrans_kernel<<<wtBlocks, 256, 0, stream>>>(Wv, WvT);
  wtrans_kernel<<<wtBlocks, 256, 0, stream>>>(Wo, WoT);

  // 3) QKV projections (V stored transposed [h*64+d][s])
  dim3 ggrid(SEQL / 128, DM / 64);
  gemm_wmma_kernel<<<ggrid, 128, 0, stream>>>(hn, WqT, bq, nullptr, qr, nullptr,
                                              SEQL, DM, DM, 0);
  gemm_wmma_kernel<<<ggrid, 128, 0, stream>>>(hn, WkT, bk, nullptr, kr, nullptr,
                                              SEQL, DM, DM, 0);
  gemm_wmma_kernel<<<ggrid, 128, 0, stream>>>(hn, WvT, bv, nullptr, vT, nullptr,
                                              SEQL, DM, DM, 1);

  // 4) RoPE in place on q,k
  rope_kernel<<<(SEQL * NH * 32) / 256, 256, 0, stream>>>(qr, kr);

  // 5) Flash attention
  dim3 agrid(SEQL / 64, NH);
  attn_kernel<<<agrid, 128, 0, stream>>>(qr, kr, vT, ob);

  // 6) Output projection + bias + residual -> f32 d_out
  gemm_wmma_kernel<<<ggrid, 128, 0, stream>>>(ob, WoT, bo, h, nullptr, out,
                                              SEQL, DM, DM, 0);
}
